// DiffFOOOFLoss_39831526703165
// MI455X (gfx1250) — compile-verified, matrix operand order checked by reading, not run on hardware
//
#include <hip/hip_runtime.h>

typedef __attribute__((ext_vector_type(2)))  float    v2f;
typedef __attribute__((ext_vector_type(8)))  float    v8f;
typedef __attribute__((ext_vector_type(16))) _Float16 v16h;

#define BLK    256
#define GRID_A 1024
#define KMAX   8

#if __has_builtin(__builtin_amdgcn_wmma_f32_16x16x4_f32)
#define HAVE_WMMA_F32X4 1
#endif

__device__ __forceinline__ float huber1(float p, float t) {
    float e = p - t;
    float a = fabsf(e);
    return (a < 1.0f) ? 0.5f * e * e : (a - 0.5f);
}

// ---------------------------------------------------------------------------
// Kernel A: Huber reduction over pred/true PSD via WMMA f32 accumulation.
// Main loop is fully unguarded (no EXEC manipulation); a single uniform tail
// iteration handles the remainder with all 32 lanes still reaching the WMMA
// (EXEC all-ones requirement). Each WMMA folds 64 f32 partials (2/lane) into
// the 8-VGPR f32 accumulator via B = ones(4x16).
// ---------------------------------------------------------------------------
__global__ __launch_bounds__(BLK) void huber_wmma_kernel(
    const float* __restrict__ pred, const float* __restrict__ truep,
    float* __restrict__ partial, long n) {
    const long nvec   = n >> 2;
    const long stride = (long)gridDim.x * blockDim.x;
    const long idx0   = (long)blockIdx.x * blockDim.x + threadIdx.x;
    const float4* __restrict__ p4 = (const float4*)pred + idx0;
    const float4* __restrict__ t4 = (const float4*)truep + idx0;

    v8f acc = {0.f, 0.f, 0.f, 0.f, 0.f, 0.f, 0.f, 0.f};
#ifdef HAVE_WMMA_F32X4
    v2f onesB;
    onesB.x = 1.0f; onesB.y = 1.0f;
#else
    v16h onesB, a16;
    for (int i = 0; i < 16; ++i) { onesB[i] = (_Float16)1.0f; a16[i] = (_Float16)0.0f; }
#endif

    const long full = nvec / stride;       // unguarded iterations (uniform)
    for (long it = 0; it < full; ++it) {
        if (it + 1 < full) {               // uniform scalar branch
            __builtin_prefetch((const void*)(p4 + stride), 0, 1);
            __builtin_prefetch((const void*)(t4 + stride), 0, 1);
        }
        float4 a = *p4;
        float4 b = *t4;
        p4 += stride;
        t4 += stride;
        float h0 = huber1(a.x, b.x), h1 = huber1(a.y, b.y);
        float h2 = huber1(a.z, b.z), h3 = huber1(a.w, b.w);
#ifdef HAVE_WMMA_F32X4
        v2f A;
        A.x = h0 + h2; A.y = h1 + h3;
        acc = __builtin_amdgcn_wmma_f32_16x16x4_f32(
            false, A, false, onesB, (short)0, acc, false, false);
#else
        a16[0] = (_Float16)h0; a16[1] = (_Float16)h1;
        a16[2] = (_Float16)h2; a16[3] = (_Float16)h3;
        acc = __builtin_amdgcn_wmma_f32_16x16x32_f16(
            false, a16, false, onesB, (short)0, acc, false, false);
#endif
    }

    // Uniform tail iteration: rem lanes (< stride) still have one vec4 left.
    const long rem = nvec - full * stride;
    if (rem > 0) {                          // uniform condition
        float h0 = 0.f, h1 = 0.f, h2 = 0.f, h3 = 0.f;
        if (idx0 < rem) {                   // divergence only around loads
            float4 a = *p4;
            float4 b = *t4;
            h0 = huber1(a.x, b.x); h1 = huber1(a.y, b.y);
            h2 = huber1(a.z, b.z); h3 = huber1(a.w, b.w);
        }
#ifdef HAVE_WMMA_F32X4
        v2f A;
        A.x = h0 + h2; A.y = h1 + h3;
        acc = __builtin_amdgcn_wmma_f32_16x16x4_f32(
            false, A, false, onesB, (short)0, acc, false, false);
#else
        a16[0] = (_Float16)h0; a16[1] = (_Float16)h1;
        a16[2] = (_Float16)h2; a16[3] = (_Float16)h3;
        acc = __builtin_amdgcn_wmma_f32_16x16x32_f16(
            false, a16, false, onesB, (short)0, acc, false, false);
#endif
    }

    // D layout (16x16 f32): VGPR r, lanes 0-15 -> M=r; lanes 16-31 -> M=8+r.
    // Sum of the 8 accumulator VGPRs gives rows 0..7 (lanes 0-15) / rows 8..15
    // (lanes 16-31); xor-16 shuffle yields the full wave sum in all lanes.
    float s = acc[0] + acc[1] + acc[2] + acc[3] + acc[4] + acc[5] + acc[6] + acc[7];
    s += __shfl_xor(s, 16, 32);

    // scalar tail (n % 4), handled once by global thread 0
    if (idx0 == 0) {
        for (long i = nvec << 2; i < n; ++i) s += huber1(pred[i], truep[i]);
    }

    __shared__ float wsum[BLK / 32];
    int lane = threadIdx.x & 31, wid = threadIdx.x >> 5;
    if (lane == 0) wsum[wid] = s;
    __syncthreads();
    if (threadIdx.x == 0) {
        float r = 0.f;
        for (int w = 0; w < BLK / 32; ++w) r += wsum[w];
        partial[blockIdx.x] = r;
    }
}

// ---------------------------------------------------------------------------
// Kernel B: per-row greedy peak matching + small loss terms. One thread per
// batch row; 7 block-reduced partial sums:
//  0: sum(amps)  1: sum(max(bws-4,0)^2)  2: (exp-gt)^2+(off-gt)^2
//  3: sum(mask)  4: matched-peaks SSE    5: sum(unmatched)  6: sum(amps*unmatched)
// ---------------------------------------------------------------------------
__global__ __launch_bounds__(BLK) void peaks_kernel(
    const float* __restrict__ cfs, const float* __restrict__ amps,
    const float* __restrict__ bws, const float* __restrict__ expo,
    const float* __restrict__ offs, const float* __restrict__ gcfs,
    const float* __restrict__ gamps, const float* __restrict__ gbws,
    const float* __restrict__ goff, const float* __restrict__ gexp,
    const float* __restrict__ mask, float* __restrict__ partial,
    int Bn, int Kn) {
    int b = blockIdx.x * blockDim.x + threadIdx.x;
    float v[7];
    for (int q = 0; q < 7; ++q) v[q] = 0.f;

    if (b < Bn) {
        float c[KMAX], a[KMAX], w[KMAX], gc[KMAX], ga[KMAX], gw[KMAX], m[KMAX];
        for (int k = 0; k < Kn; ++k) {
            c[k]  = cfs[b * Kn + k];  a[k]  = amps[b * Kn + k];  w[k]  = bws[b * Kn + k];
            gc[k] = gcfs[b * Kn + k]; ga[k] = gamps[b * Kn + k]; gw[k] = gbws[b * Kn + k];
            m[k]  = mask[b * Kn + k];
        }
        bool used[KMAX];
        for (int k = 0; k < Kn; ++k) used[k] = false;

        float speaks = 0.f;
        for (int j = 0; j < Kn; ++j) {              // sequential over GT slots
            bool act = m[j] > 0.5f;
            float best = 3.0e38f; int bi = 0;
            for (int k = 0; k < Kn; ++k) {          // argmin, first-wins on ties
                float d = used[k] ? 1.0e9f : fabsf(gc[j] - c[k]);
                if (d < best) { best = d; bi = k; }
            }
            if (act) {
                used[bi] = true;
                float dc = c[bi] - gc[j], da = a[bi] - ga[j], dw = w[bi] - gw[j];
                speaks += dc * dc + da * da + dw * dw;
            }
        }
        float samps = 0.f, sbw = 0.f, smask = 0.f, sunm = 0.f, sum_ = 0.f;
        for (int k = 0; k < Kn; ++k) {
            samps += a[k];
            float ex = fmaxf(w[k] - 4.0f, 0.f);
            sbw += ex * ex;
            smask += m[k];
            float un = used[k] ? 0.f : 1.f;
            sunm += un;
            sum_ += a[k] * un;
        }
        float de = expo[b] - gexp[b], dofs = offs[b] - goff[b];
        v[0] = samps; v[1] = sbw; v[2] = de * de + dofs * dofs;
        v[3] = smask; v[4] = speaks; v[5] = sunm; v[6] = sum_;
    }

    __shared__ float red[BLK];
    for (int q = 0; q < 7; ++q) {
        red[threadIdx.x] = v[q];
        __syncthreads();
        for (int s = BLK / 2; s > 0; s >>= 1) {
            if (threadIdx.x < s) red[threadIdx.x] += red[threadIdx.x + s];
            __syncthreads();
        }
        if (threadIdx.x == 0) partial[q * gridDim.x + blockIdx.x] = red[0];
        __syncthreads();
    }
}

// ---------------------------------------------------------------------------
// Kernel C: single block folds all partials into the final scalar loss.
// ---------------------------------------------------------------------------
__global__ __launch_bounds__(BLK) void finalize_kernel(
    const float* __restrict__ hpart, int nA,
    const float* __restrict__ spart, int nB,
    float* __restrict__ out, float n_psd_inv, float b_inv, float bk_inv) {
    __shared__ float red[BLK];
    float s = 0.f;
    for (int i = threadIdx.x; i < nA; i += BLK) s += hpart[i];
    red[threadIdx.x] = s;
    __syncthreads();
    for (int t = BLK / 2; t > 0; t >>= 1) {
        if (threadIdx.x < t) red[threadIdx.x] += red[threadIdx.x + t];
        __syncthreads();
    }
    if (threadIdx.x == 0) {
        float hs = red[0];
        float q[7];
        for (int qq = 0; qq < 7; ++qq) {
            float a = 0.f;
            for (int i = 0; i < nB; ++i) a += spart[qq * nB + i];
            q[qq] = a;
        }
        float l_recon  = hs * n_psd_inv;
        float l_sparse = q[0] * bk_inv;
        float l_bw     = q[1] * bk_inv;
        float l_ap     = q[2] * b_inv;
        float l_peaks  = q[4] / fmaxf(q[3], 1.0f);
        float l_um     = q[6] / fmaxf(q[5], 1.0f);
        out[0] = l_recon + 0.1f * l_sparse + 0.05f * l_bw +
                 0.5f * l_ap + 0.3f * l_peaks + 0.1f * l_um;
    }
}

extern "C" void kernel_launch(void* const* d_in, const int* in_sizes, int n_in,
                              void* d_out, int out_size, void* d_ws, size_t ws_size,
                              hipStream_t stream) {
    const float* pred  = (const float*)d_in[0];
    const float* truep = (const float*)d_in[1];
    const float* cfs   = (const float*)d_in[2];
    const float* amps  = (const float*)d_in[3];
    const float* bws   = (const float*)d_in[4];
    const float* expo  = (const float*)d_in[5];
    const float* offs  = (const float*)d_in[6];
    const float* gcfs  = (const float*)d_in[7];
    const float* gamps = (const float*)d_in[8];
    const float* gbws  = (const float*)d_in[9];
    const float* goff  = (const float*)d_in[10];
    const float* gexp  = (const float*)d_in[11];
    const float* mask  = (const float*)d_in[12];

    long n_psd = in_sizes[0];
    int  Bn    = in_sizes[5];            // exponent is [B, 1]
    int  Kn    = in_sizes[2] / Bn;       // cfs is [B, K]
    if (Kn > KMAX) Kn = KMAX;

    float* ws    = (float*)d_ws;
    float* hpart = ws;                   // GRID_A floats
    float* spart = ws + GRID_A;          // 7 * gridB floats
    int gridB = (Bn + BLK - 1) / BLK;

    huber_wmma_kernel<<<GRID_A, BLK, 0, stream>>>(pred, truep, hpart, n_psd);
    peaks_kernel<<<gridB, BLK, 0, stream>>>(cfs, amps, bws, expo, offs,
                                            gcfs, gamps, gbws, goff, gexp, mask,
                                            spart, Bn, Kn);
    finalize_kernel<<<1, BLK, 0, stream>>>(hpart, GRID_A, spart, gridB,
                                           (float*)d_out,
                                           1.0f / (float)n_psd,
                                           1.0f / (float)Bn,
                                           1.0f / (float)(Bn * Kn));
}
